// MinLSTM_32916629356933
// MI455X (gfx1250) — matrix operand order repaired and checked by reading, not compiled, not run on hardware
//
#include <hip/hip_runtime.h>

// ---------------------------------------------------------------------------
// MinLSTM fused kernel set for MI455X (gfx1250, wave32, WMMA bf16 + TDM)
//   pass 0: f32 -> bf16 conversion of x and w into workspace
//   pass 1: g = xb @ wb (M=32768, K=1024, N=3*1024) with A tiles staged by
//           the Tensor Data Mover (tensor_load_to_lds, TENSORcnt), gates fused
//   pass 2-4: chunked parallel scan of cumsum(log_f_prime) along T
// ---------------------------------------------------------------------------

#define B_DIM 8
#define T_DIM 4096
#define H_DIM 1024
#define M_DIM (B_DIM * T_DIM)         // 32768
#define EPSV 1e-8f
#define LOG_EPS -18.420680743952367f  // log(1e-8)

#define BM 128      // macro-tile rows
#define BN 32       // macro-tile logical columns (within each H group)
#define KT 32       // K tile
#define LDSP 40     // padded K stride in LDS (bf16 elems): 64B row + 16B pad
#define NCH 16      // scan chunks
#define TC (T_DIM / NCH)  // 256 timesteps per chunk

typedef __attribute__((ext_vector_type(16))) __bf16 v16bf;
typedef __attribute__((ext_vector_type(8)))  __bf16 v8bf;
typedef __attribute__((ext_vector_type(8)))  float  v8f;
typedef __attribute__((ext_vector_type(4))) unsigned int u32x4;
typedef __attribute__((ext_vector_type(8))) int          i32x8;
typedef __attribute__((ext_vector_type(4))) int          i32x4;

union V16U { v16bf v; v8bf h[2]; };

__device__ __forceinline__ unsigned short f2bf(float f) {
    unsigned int u = __float_as_uint(f);
    u = (u + 0x7FFFu + ((u >> 16) & 1u)) >> 16;   // round-to-nearest-even
    return (unsigned short)u;
}

__device__ __forceinline__ float log_add_exp(float a, float b) {
    float m  = fmaxf(a, b);
    float mn = fminf(a, b);
    return m + __logf(1.0f + __expf(mn - m));
}

// ---------------------------------------------------------------------------
// Kernel 0: one-shot f32 -> bf16 conversion of x (33.5M) and w (3.1M)
// ---------------------------------------------------------------------------
#define XN4 ((M_DIM * H_DIM) / 4)            // 8388608 float4 in x
#define WN4 ((H_DIM * 3 * H_DIM) / 4)        // 786432  float4 in w
__global__ __launch_bounds__(256)
void minlstm_convert(const float* __restrict__ x, const float* __restrict__ w,
                     unsigned short* __restrict__ xb, unsigned short* __restrict__ wb) {
    unsigned int i = blockIdx.x * 256 + threadIdx.x;
    const float* src;
    uint2* dst;
    if (i < XN4) { src = x + (size_t)i * 4;         dst = (uint2*)xb + i; }
    else { unsigned int j = i - XN4; src = w + (size_t)j * 4; dst = (uint2*)wb + j; }
    float4 v = *(const float4*)src;
    uint2 p;
    p.x = (unsigned int)f2bf(v.x) | ((unsigned int)f2bf(v.y) << 16);
    p.y = (unsigned int)f2bf(v.z) | ((unsigned int)f2bf(v.w) << 16);
    *dst = p;
}

// ---------------------------------------------------------------------------
// Kernel 1: fused GEMM (bf16 WMMA, f32 accum) + gate math.
//   A tile (128x32 bf16) staged by the Tensor Data Mover with in-flight row
//   padding (64B data + 16B pad -> LDSP=40). B tiles staged manually
//   (transposed to [n][k] for contiguous per-lane K).
//   writes log_f_prime -> lfp[],  log_state -> out[] (overwritten by scan)
// ---------------------------------------------------------------------------
__global__ __launch_bounds__(256)
void minlstm_gemm_gates(const unsigned short* __restrict__ xb,
                        const unsigned short* __restrict__ wb,
                        float* __restrict__ lfp,
                        float* __restrict__ out) {
    __shared__ __attribute__((aligned(16))) unsigned short Alds[BM * LDSP];
    __shared__ __attribute__((aligned(16))) unsigned short Blds[3 * BN * LDSP];

    const int tid    = threadIdx.x;
    const int lane   = tid & 31;
    const int wid    = tid >> 5;       // 0..7
    const int wave_m = wid & 3;        // 4 waves along M (32 rows each)
    const int wave_n = wid >> 2;       // 2 waves along N (16 cols each)
    const int half   = lane >> 4;      // 0 | 1
    const int l16    = lane & 15;

    const int row0 = blockIdx.x * BM;  // M offset
    const int n0   = blockIdx.y * BN;  // logical column offset within H

    // Tensor DMA descriptor constants (ISA ch.8): 2D bf16 tile 32(K) x 128(M)
    // group1: data_size=2B, pad_enable, pad_interval=16DW(64B), pad_amount=4DW(16B)
    const unsigned int g1d0 = 0x00010000u | (1u << 20) | (3u << 22) | (3u << 25);
    const i32x8 g1 = { (int)g1d0,
                       (int)0x04000000u,          // tensor_dim0 = 1024 (lo16 @ bits 63:48)
                       (int)(128u << 16),         // tensor_dim1 = 128  (lo16 @ bits 95:80)
                       (int)(32u << 16),          // tile_dim0 = 32     (bits 127:112)
                       128,                       // tile_dim1 = 128    (bits 143:128)
                       1024,                      // tensor_dim0_stride = 1024 (bits 191:160)
                       0, 0 };
    const i32x4 zg4 = { 0, 0, 0, 0 };
    const i32x8 zg8 = { 0, 0, 0, 0, 0, 0, 0, 0 };
    const unsigned int ldsA = (unsigned int)(uintptr_t)&Alds[0];

    const uint2* wb2 = (const uint2*)wb;   // 4 bf16 per element

    // precomputed per-thread B-staging coordinates (j = 0..2, 768 uint2 total)
    int bg[3], bk[3], bnq[3];
#pragma unroll
    for (int j = 0; j < 3; ++j) {
        int idx = tid + j * 256;
        bg[j]  = idx >> 8;                 // gate group 0..2
        int rem = idx & 255;
        bk[j]  = rem >> 3;                 // K row 0..31
        bnq[j] = (rem & 7) << 2;           // starting n (groups of 4)
    }

    v8f accF[2], accI[2], accH[2];
    const v8f vzero = {0.f, 0.f, 0.f, 0.f, 0.f, 0.f, 0.f, 0.f};
#pragma unroll
    for (int mt = 0; mt < 2; ++mt) { accF[mt] = vzero; accI[mt] = vzero; accH[mt] = vzero; }

    for (int k0 = 0; k0 < H_DIM; k0 += KT) {
        // ---- A tile via Tensor Data Mover: global bf16 -> padded LDS ----
        if (wid == 0) {
            unsigned long long ga = (unsigned long long)(uintptr_t)xb
                                  + ((unsigned long long)row0 * H_DIM + (unsigned)k0) * 2ull;
            u32x4 g0 = { 1u,                                   // count=1, load, no gather
                         ldsA,                                 // lds_addr
                         (unsigned int)ga,                     // global_addr[31:0]
                         ((unsigned int)(ga >> 32) & 0x1FFFFFFu) | (2u << 30) }; // addr[56:32] | type=2
            __builtin_amdgcn_tensor_load_to_lds(g0, g1, zg4, zg4, zg8, 0);
        }
        // ---- stage B tiles: 3 groups x 32K x 32N bf16, transposed to [n][k] ----
        // phase 1: issue all three b64 loads into independent registers
        uint2 bv[3];
#pragma unroll
        for (int j = 0; j < 3; ++j) {
            bv[j] = wb2[(size_t)(k0 + bk[j]) * (3 * H_DIM / 4) + bg[j] * (H_DIM / 4)
                        + ((n0 + bnq[j]) >> 2)];
        }
        // phase 2: scatter into transposed LDS layout
#pragma unroll
        for (int j = 0; j < 3; ++j) {
            unsigned short* bb = &Blds[bg[j] * BN * LDSP + bk[j]];
            int nq = bnq[j];
            bb[(nq + 0) * LDSP] = (unsigned short)(bv[j].x & 0xFFFFu);
            bb[(nq + 1) * LDSP] = (unsigned short)(bv[j].x >> 16);
            bb[(nq + 2) * LDSP] = (unsigned short)(bv[j].y & 0xFFFFu);
            bb[(nq + 3) * LDSP] = (unsigned short)(bv[j].y >> 16);
        }
        // prefetch next w K-tile while this one is consumed
        if (k0 + KT < H_DIM && tid < 96) {
            int kk = tid & 31, seg = tid >> 5;
            __builtin_prefetch(wb + (size_t)(k0 + KT + kk) * (3 * H_DIM) + seg * H_DIM + n0, 0, 0);
        }
        if (wid == 0) __builtin_amdgcn_s_wait_tensorcnt(0);   // TDM drain (issuing wave)
        __syncthreads();

        // ---- B fragments (one per gate group) ----
        V16U bfr[3];
#pragma unroll
        for (int g = 0; g < 3; ++g) {
            const unsigned short* bp = &Blds[g * BN * LDSP + (wave_n * 16 + l16) * LDSP];
            bfr[g].h[0] = *(const v8bf*)(bp + half * 8);
            bfr[g].h[1] = *(const v8bf*)(bp + 16 + half * 8);
        }
        // ---- A fragments + WMMA ----
#pragma unroll
        for (int mt = 0; mt < 2; ++mt) {
            const unsigned short* ap = &Alds[(wave_m * 32 + mt * 16 + l16) * LDSP];
            V16U af;
            af.h[0] = *(const v8bf*)(ap + half * 8);        // K 0..7  / 8..15
            af.h[1] = *(const v8bf*)(ap + 16 + half * 8);   // K 16..23 / 24..31
            accF[mt] = __builtin_amdgcn_wmma_f32_16x16x32_bf16(
                false, af.v, false, bfr[0].v, (short)0, accF[mt], false, false);
            accI[mt] = __builtin_amdgcn_wmma_f32_16x16x32_bf16(
                false, af.v, false, bfr[1].v, (short)0, accI[mt], false, false);
            accH[mt] = __builtin_amdgcn_wmma_f32_16x16x32_bf16(
                false, af.v, false, bfr[2].v, (short)0, accH[mt], false, false);
        }
        __syncthreads();
    }

    // ---- fused gate epilogue ----
    const int col = n0 + wave_n * 16 + l16;             // column within H
#pragma unroll
    for (int mt = 0; mt < 2; ++mt) {
        const int rbase = row0 + wave_m * 32 + mt * 16 + half * 8;
#pragma unroll
        for (int v = 0; v < 8; ++v) {
            float gf = accF[mt][v];
            float gi = accI[mt][v];
            float th = accH[mt][v];
            float fg = 1.0f / (1.0f + __expf(-gf));
            float ig = 1.0f / (1.0f + __expf(-gi));
            float log_f = __logf(fg + EPSV);
            float log_i = __logf(ig + EPSV);
            float lse = log_add_exp(log_f, log_i);
            float log_f_prime = lse - log_f;
            float log_i_prime = lse - log_i;
            float log_state = log_add_exp(log_f_prime + LOG_EPS,
                                          log_i_prime + __logf(th + EPSV));
            size_t o = (size_t)(rbase + v) * H_DIM + col;
            lfp[o] = log_f_prime;
            out[o] = log_state;
        }
    }
}

// ---------------------------------------------------------------------------
// Kernel 2: per-chunk sums of log_f_prime   (131072 threads)
// ---------------------------------------------------------------------------
__global__ __launch_bounds__(256)
void minlstm_scan_sum(const float* __restrict__ lfp, float* __restrict__ part) {
    int g  = blockIdx.x * 256 + threadIdx.x;       // B*NCH*H = 131072
    int h  = g & (H_DIM - 1);
    int ch = (g >> 10) & (NCH - 1);
    int b  = g >> 14;
    float s = 0.0f;
    size_t base = ((size_t)b * T_DIM + ch * TC) * H_DIM + h;
    for (int t = 0; t < TC; ++t) s += lfp[base + (size_t)t * H_DIM];
    part[((size_t)b * NCH + ch) * H_DIM + h] = s;
}

// ---------------------------------------------------------------------------
// Kernel 3: exclusive prefix over the 16 chunk sums per (b,h)
// ---------------------------------------------------------------------------
__global__ __launch_bounds__(256)
void minlstm_scan_prefix(float* __restrict__ part) {
    int g = blockIdx.x * 256 + threadIdx.x;        // B*H = 8192
    int h = g & (H_DIM - 1);
    int b = g >> 10;
    float run = 0.0f;
#pragma unroll
    for (int ch = 0; ch < NCH; ++ch) {
        size_t idx = ((size_t)b * NCH + ch) * H_DIM + h;
        float s = part[idx];
        part[idx] = run;
        run += s;
    }
}

// ---------------------------------------------------------------------------
// Kernel 4: rescan chunk, out = exp(cumsum(log_f_prime) + log_state)
// ---------------------------------------------------------------------------
__global__ __launch_bounds__(256)
void minlstm_scan_apply(const float* __restrict__ lfp,
                        const float* __restrict__ part,
                        float* __restrict__ out) {
    int g  = blockIdx.x * 256 + threadIdx.x;
    int h  = g & (H_DIM - 1);
    int ch = (g >> 10) & (NCH - 1);
    int b  = g >> 14;
    float cum = part[((size_t)b * NCH + ch) * H_DIM + h];
    size_t base = ((size_t)b * T_DIM + ch * TC) * H_DIM + h;
    for (int t = 0; t < TC; ++t) {
        size_t idx = base + (size_t)t * H_DIM;
        cum += lfp[idx];
        out[idx] = __expf(cum + out[idx]);   // out held log_state
    }
}

// ---------------------------------------------------------------------------
extern "C" void kernel_launch(void* const* d_in, const int* in_sizes, int n_in,
                              void* d_out, int out_size, void* d_ws, size_t ws_size,
                              hipStream_t stream) {
    const float* x = (const float*)d_in[0];   // (B,T,H) f32
    const float* w = (const float*)d_in[1];   // (H,3H)  f32
    float* out = (float*)d_out;               // (B,T,H) f32

    // workspace layout
    float*          lfp  = (float*)d_ws;                          // M*H f32
    float*          part = lfp + (size_t)M_DIM * H_DIM;           // B*NCH*H f32
    unsigned short* xbuf = (unsigned short*)(part + (size_t)B_DIM * NCH * H_DIM);
    unsigned short* wbuf = xbuf + (size_t)M_DIM * H_DIM;

    minlstm_convert<<<(XN4 + WN4) / 256, 256, 0, stream>>>(x, w, xbuf, wbuf);

    dim3 ggrid(M_DIM / BM, H_DIM / BN);       // 256 x 32 workgroups
    minlstm_gemm_gates<<<ggrid, 256, 0, stream>>>(xbuf, wbuf, lfp, out);

    minlstm_scan_sum<<<(B_DIM * NCH * H_DIM) / 256, 256, 0, stream>>>(lfp, part);
    minlstm_scan_prefix<<<(B_DIM * H_DIM) / 256, 256, 0, stream>>>(part);
    minlstm_scan_apply<<<(B_DIM * NCH * H_DIM) / 256, 256, 0, stream>>>(lfp, part, out);
}
